// DecoderWrapper_11287174054670
// MI455X (gfx1250) — matrix operand order repaired and checked by reading, not compiled, not run on hardware
//
#include <hip/hip_runtime.h>

// ---------------------------------------------------------------------------
// LSTM decoder w/ Luong attention for MI455X (gfx1250).
// Weights converted once to bf16 (~41MB, L2-resident at 192MB), all GEMMs on
// v_wmma_f32_16x16x32_bf16 with fp32 accumulation. M=64 makes every GEMM
// weight-bandwidth bound, so bf16 halves the dominant byte stream. Recurrence
// is a host loop of 6 lean kernels per step on `stream` (graph-capture safe).
// ---------------------------------------------------------------------------

typedef __attribute__((ext_vector_type(16))) __bf16 v16bf;
typedef __attribute__((ext_vector_type(8)))  __bf16 v8bf;
typedef __attribute__((ext_vector_type(8)))  float  v8f;

#define BB 64
#define TT 48
#define SS 48
#define EE 512
#define HH 1024
#define LL 2
#define GG 4096   // 4*H

#if __has_builtin(__builtin_amdgcn_tanhf)
__device__ __forceinline__ float fast_tanh(float x) { return __builtin_amdgcn_tanhf(x); }
#else
__device__ __forceinline__ float fast_tanh(float x) { return tanhf(x); }
#endif

__device__ __forceinline__ float sigmoidf_(float x) {
    return 1.f / (1.f + __expf(-x));
}

// A-fragment (16x32 bf16, wave32): lanes 0-15 hold row M=lane, K {k0..k0+15};
// lanes 16-31 hold row M=lane-16, K {k0+16..k0+31}.
__device__ __forceinline__ v16bf load_a_frag(const __bf16* __restrict__ row,
                                             int kk, bool hi) {
    int off = kk + (hi ? 16 : 0);
    v8bf a0 = *(const v8bf*)(row + off);
    v8bf a1 = *(const v8bf*)(row + off + 8);
    return __builtin_shufflevector(a0, a1, 0, 1, 2, 3, 4, 5, 6, 7,
                                   8, 9, 10, 11, 12, 13, 14, 15);
}

// out[64, N] = concat(Xa[:, :Ka], Xb[:, :Kb]) @ W1^T (+ Hc @ W2^T) + biases.
// W row-major [N, K]; B-fragment: lane n reads 16 contiguous bf16 (32B) from
// weight row (nTile*16 + n%16) at column k0 + (lane>=16 ? 16 : 0).
// One wave per 16x16 D tile: wid -> (nTile = wid/4, mTile = wid%4).
// Waves 0-3 (and 4-7) of a block share an nTile, so the B stream is
// deduplicated by WGP$; L2 sees each weight element once per step.
__global__ void __launch_bounds__(256)
wmma_gemm_kernel(float* __restrict__ out, int N,
                 const __bf16* __restrict__ Xa, int ldxa, int Ka,
                 const __bf16* __restrict__ Xb, int ldxb, int Kb,
                 const __bf16* __restrict__ W1,
                 const __bf16* __restrict__ Hc, int Kh,
                 const __bf16* __restrict__ W2,
                 const float* __restrict__ bias1,
                 const float* __restrict__ bias2) {
    const int lane = threadIdx.x & 31;
    const bool hi  = lane >= 16;
    const int n16  = lane & 15;
    const int wid  = blockIdx.x * 8 + (threadIdx.x >> 5);
    const int nTile = wid >> 2;
    const int mTile = wid & 3;
    if (nTile >= (N >> 4)) return;   // uniform per-wave; EXEC stays all-ones

    const int m  = mTile * 16 + n16;       // A row this lane feeds
    const int K1 = Ka + Kb;
    const __bf16* w1row = W1 + (size_t)(nTile * 16 + n16) * K1 + (hi ? 16 : 0);
    v8f acc = {};

    {   // first A segment (tight loop: no per-iteration source select)
        const __bf16* arow = Xa + (size_t)m * ldxa;
        for (int k0 = 0; k0 < Ka; k0 += 32) {
            __builtin_prefetch(w1row + k0 + 128, 0, 1);   // global_prefetch_b8
            v16bf af = load_a_frag(arow, k0, hi);
            v16bf bf = *(const v16bf*)(w1row + k0);
            acc = __builtin_amdgcn_wmma_f32_16x16x32_bf16(
                      false, af, false, bf, (short)0, acc, false, false);
        }
    }
    if (Xb) {   // second A segment of the concat (same W1, column offset Ka)
        const __bf16* arow = Xb + (size_t)m * ldxb;
        const __bf16* wseg = w1row + Ka;
        for (int k0 = 0; k0 < Kb; k0 += 32) {
            __builtin_prefetch(wseg + k0 + 128, 0, 1);
            v16bf af = load_a_frag(arow, k0, hi);
            v16bf bf = *(const v16bf*)(wseg + k0);
            acc = __builtin_amdgcn_wmma_f32_16x16x32_bf16(
                      false, af, false, bf, (short)0, acc, false, false);
        }
    }
    if (Hc) {   // recurrent term: Hc @ W2^T
        const __bf16* w2row = W2 + (size_t)(nTile * 16 + n16) * Kh + (hi ? 16 : 0);
        const __bf16* arow  = Hc + (size_t)m * Kh;
        for (int k0 = 0; k0 < Kh; k0 += 32) {
            __builtin_prefetch(w2row + k0 + 128, 0, 1);
            v16bf af = load_a_frag(arow, k0, hi);
            v16bf bf = *(const v16bf*)(w2row + k0);
            acc = __builtin_amdgcn_wmma_f32_16x16x32_bf16(
                      false, af, false, bf, (short)0, acc, false, false);
        }
    }
    // D layout: VGPR r -> M = r (lanes 0-15), M = 8+r (lanes 16-31); N = lane%16
    const int col  = nTile * 16 + n16;
    const float bs = (bias1 ? bias1[col] : 0.f) + (bias2 ? bias2[col] : 0.f);
    const int rbase = mTile * 16 + (hi ? 8 : 0);
#pragma unroll
    for (int r = 0; r < 8; ++r)
        out[(size_t)(rbase + r) * N + col] = acc[r] + bs;
}

// Luong attention: one workgroup per batch row b.
__global__ void __launch_bounds__(256)
attn_kernel(const float* __restrict__ q,        // [B,H]
            const float* __restrict__ enc,      // [B,S,H]
            const unsigned char* __restrict__ mask, // [B,S] (bool)
            __bf16* __restrict__ ctxb) {        // [B,H] bf16 out
    const int b = blockIdx.x;
    __shared__ float sc[SS];
    __shared__ float wgt[SS];
    const int wave = threadIdx.x >> 5, lane = threadIdx.x & 31;
    const float* qr = q + (size_t)b * HH;
    for (int s = wave; s < SS; s += 8) {
        const float* er = enc + ((size_t)b * SS + s) * HH;
        float p = 0.f;
        for (int k = lane; k < HH; k += 32) p = __fmaf_rn(qr[k], er[k], p);
        for (int off = 16; off; off >>= 1) p += __shfl_xor(p, off, 32);
        if (lane == 0) sc[s] = mask[b * SS + s] ? -3.0e38f : p;
    }
    __syncthreads();
    if (threadIdx.x < SS) {
        float mx = -3.0e38f;
        for (int s = 0; s < SS; ++s) mx = fmaxf(mx, sc[s]);
        float sum = 0.f;
        for (int s = 0; s < SS; ++s) sum += __expf(sc[s] - mx);
        wgt[threadIdx.x] = __expf(sc[threadIdx.x] - mx) / sum;
    }
    __syncthreads();
    for (int hh = threadIdx.x; hh < HH; hh += 256) {
        float acc = 0.f;
        for (int s = 0; s < SS; ++s)
            acc = __fmaf_rn(wgt[s], enc[((size_t)b * SS + s) * HH + hh], acc);
        ctxb[(size_t)b * HH + hh] = (__bf16)acc;
    }
}

// Pointwise LSTM cell update (PyTorch gate order i,f,g,o).
__global__ void __launch_bounds__(256)
lstm_kernel(const float* __restrict__ gates,    // [B,4H]
            float* __restrict__ h, float* __restrict__ c, // [B,H] fp32 state
            __bf16* __restrict__ hb,            // bf16 mirror of h
            float* __restrict__ outp) {         // optional: outputs + t*H
    int i = blockIdx.x * 256 + threadIdx.x;
    if (i >= BB * HH) return;
    int b = i / HH, hh = i % HH;
    const float* g = gates + (size_t)b * GG;
    float ig = sigmoidf_(g[hh]);
    float fg = sigmoidf_(g[HH + hh]);
    float gg = fast_tanh(g[2 * HH + hh]);
    float og = sigmoidf_(g[3 * HH + hh]);
    float cn = fg * c[i] + ig * gg;
    float hn = og * fast_tanh(cn);
    c[i] = cn;
    h[i] = hn;
    hb[i] = (__bf16)hn;
    if (outp) outp[(size_t)b * TT * HH + hh] = hn;
}

__global__ void __launch_bounds__(256)
cvt_kernel(const float* __restrict__ src, __bf16* __restrict__ dst, int n) {
    int i = blockIdx.x * 256 + threadIdx.x;
    if (i < n) dst[i] = (__bf16)src[i];
}

__global__ void __launch_bounds__(256)
embed_kernel(const int* __restrict__ tgt, const float* __restrict__ emb,
             __bf16* __restrict__ out) {        // out [T,B,E]
    int i = blockIdx.x * 256 + threadIdx.x;
    if (i >= TT * BB * EE) return;
    int t = i / (BB * EE);
    int r = i % (BB * EE);
    int b = r / EE, e = r % EE;
    int tok = tgt[b * TT + t];
    out[i] = (__bf16)emb[(size_t)tok * EE + e];
}

__global__ void __launch_bounds__(256)
initstate_kernel(const float* __restrict__ h0, const float* __restrict__ c0,
                 float* __restrict__ hs, float* __restrict__ cs,
                 __bf16* __restrict__ h0b, __bf16* __restrict__ h1b) {
    int i = blockIdx.x * 256 + threadIdx.x;
    if (i >= LL * BB * HH) return;
    hs[i] = h0[i];
    cs[i] = c0[i];
    if (i < BB * HH) h0b[i] = (__bf16)h0[i];
    else             h1b[i - BB * HH] = (__bf16)h0[i];
}

extern "C" void kernel_launch(void* const* d_in, const int* in_sizes, int n_in,
                              void* d_out, int out_size, void* d_ws, size_t ws_size,
                              hipStream_t stream) {
    const int*   tgt   = (const int*)d_in[0];
    const float* h0    = (const float*)d_in[1];
    const float* c0    = (const float*)d_in[2];
    const float* enc   = (const float*)d_in[3];
    const unsigned char* mask = (const unsigned char*)d_in[4];
    const float* emb   = (const float*)d_in[5];
    const float* W_l   = (const float*)d_in[6];
    const float* b_l   = (const float*)d_in[7];
    const float* W_ih0 = (const float*)d_in[8];
    const float* W_hh0 = (const float*)d_in[9];
    const float* b_ih0 = (const float*)d_in[10];
    const float* b_hh0 = (const float*)d_in[11];
    const float* W_ih1 = (const float*)d_in[12];
    const float* W_hh1 = (const float*)d_in[13];
    const float* b_ih1 = (const float*)d_in[14];
    const float* b_hh1 = (const float*)d_in[15];
    (void)in_sizes; (void)n_in; (void)out_size;

    // d_out = outputs[B,T,H] ++ hn[L,B,H] ++ cn[L,B,H]; keep live state in-place.
    float* outputs = (float*)d_out;
    float* hstate  = outputs + (size_t)BB * TT * HH;
    float* cstate  = hstate + (size_t)LL * BB * HH;

    // workspace carve-out (~45 MB)
    char* base = (char*)d_ws;
    size_t off = 0;
    auto alloc = [&](size_t bytes) -> void* {
        void* p = base + off;
        off = (off + bytes + 511) & ~(size_t)511;
        return p;
    };
    __bf16* wWl   = (__bf16*)alloc(2ull * HH * HH);
    __bf16* wWih0 = (__bf16*)alloc(2ull * GG * (EE + HH));
    __bf16* wWhh0 = (__bf16*)alloc(2ull * GG * HH);
    __bf16* wWih1 = (__bf16*)alloc(2ull * GG * HH);
    __bf16* wWhh1 = (__bf16*)alloc(2ull * GG * HH);
    __bf16* embB  = (__bf16*)alloc(2ull * TT * BB * EE);
    __bf16* h0b   = (__bf16*)alloc(2ull * BB * HH);
    __bf16* h1b   = (__bf16*)alloc(2ull * BB * HH);
    __bf16* ctxb  = (__bf16*)alloc(2ull * BB * HH);
    float*  qbuf  = (float*)alloc(4ull * BB * HH);
    float*  gbuf  = (float*)alloc(4ull * BB * GG);
    (void)ws_size;

    // ---- prologue: bf16 weight conversion + embedding gather + state init
    auto cvt = [&](const float* s, __bf16* d, int n) {
        cvt_kernel<<<(n + 255) / 256, 256, 0, stream>>>(s, d, n);
    };
    cvt(W_l,   wWl,   HH * HH);
    cvt(W_ih0, wWih0, GG * (EE + HH));
    cvt(W_hh0, wWhh0, GG * HH);
    cvt(W_ih1, wWih1, GG * HH);
    cvt(W_hh1, wWhh1, GG * HH);
    embed_kernel<<<(TT * BB * EE + 255) / 256, 256, 0, stream>>>(tgt, emb, embB);
    initstate_kernel<<<(LL * BB * HH + 255) / 256, 256, 0, stream>>>(
        h0, c0, hstate, cstate, h0b, h1b);

    // ---- sequential decode: 6 kernels per step
    for (int t = 0; t < TT; ++t) {
        // q = h1 @ W_l^T + b_l                       (WMMA, N=1024)
        wmma_gemm_kernel<<<HH / 32, 256, 0, stream>>>(
            qbuf, HH, h1b, HH, HH, nullptr, 0, 0, wWl,
            nullptr, 0, nullptr, b_l, nullptr);
        // attention scores/softmax/context (fp32) -> ctx bf16
        attn_kernel<<<BB, 256, 0, stream>>>(qbuf, enc, mask, ctxb);
        // gates0 = [emb_t, ctx] @ W_ih0^T + h0 @ W_hh0^T + b   (WMMA, N=4096)
        wmma_gemm_kernel<<<GG / 32, 256, 0, stream>>>(
            gbuf, GG, embB + (size_t)t * BB * EE, EE, EE, ctxb, HH, HH, wWih0,
            h0b, HH, wWhh0, b_ih0, b_hh0);
        lstm_kernel<<<(BB * HH + 255) / 256, 256, 0, stream>>>(
            gbuf, hstate, cstate, h0b, nullptr);
        // gates1 = h0 @ W_ih1^T + h1 @ W_hh1^T + b             (WMMA, N=4096)
        wmma_gemm_kernel<<<GG / 32, 256, 0, stream>>>(
            gbuf, GG, h0b, HH, HH, nullptr, 0, 0, wWih1,
            h1b, HH, wWhh1, b_ih1, b_hh1);
        // layer-1 cell + write outputs[:, t, :]
        lstm_kernel<<<(BB * HH + 255) / 256, 256, 0, stream>>>(
            gbuf, hstate + (size_t)BB * HH, cstate + (size_t)BB * HH, h1b,
            outputs + (size_t)t * HH);
    }
}
